// LSTMSlidingWindow_66099546685953
// MI455X (gfx1250) — compile-verified, matrix operand order checked by reading, not run on hardware
//
#include <hip/hip_runtime.h>
#include <hip/hip_bf16.h>

typedef _Float16 v16h __attribute__((ext_vector_type(16)));
typedef _Float16 v8h  __attribute__((ext_vector_type(8)));
typedef float    v8f  __attribute__((ext_vector_type(8)));

#define TFRM 128
#define H0 480
#define W0 640
#define C1H 118
#define C1W 158
#define P1H 58
#define P1W 78
#define C2H 27
#define C2W 37
#define P2H 13
#define P2W 18
#define C3H 11
#define C3W 16
#define C4H 9
#define C4W 14
#define FEAT 576
#define HID 128
#define G4 512
#define NSTEP 738   // 123 windows * fps 6

// ---------------- Kernel 1: conv1(11x11,s4) + ReLU + maxpool(3x3,s2), fused per tile ----
__global__ void __launch_bounds__(256) k_conv1_pool1(const float* __restrict__ video,
                                                     const float* __restrict__ w,
                                                     const float* __restrict__ b,
                                                     float* __restrict__ out) {
    __shared__ float in_t[75 * 75];
    __shared__ float w_t[12 * 121];
    __shared__ float conv_t[12 * 289];
    const int f   = blockIdx.z;
    const int py0 = blockIdx.y * 8, px0 = blockIdx.x * 8;
    const int iy0 = 8 * py0, ix0 = 8 * px0;
    const float* img = video + (size_t)f * H0 * W0;
    const int tid = threadIdx.x;

    for (int i = tid; i < 12 * 121; i += 256) w_t[i] = w[i];
    for (int i = tid; i < 75 * 75; i += 256) {
        int r = i / 75, c = i % 75;
        int iy = iy0 + r, ix = ix0 + c;
        if (iy + 8 < H0 && ix < W0) __builtin_prefetch(&img[(iy + 8) * W0 + ix], 0, 0);
        in_t[i] = (iy < H0 && ix < W0) ? img[iy * W0 + ix] : 0.f;
    }
    __syncthreads();

    for (int i = tid; i < 12 * 289; i += 256) {
        int oc = i / 289, rr = (i % 289) / 17, cc = i % 17;
        const float* wp = w_t + oc * 121;
        const float* ip = in_t + (4 * rr) * 75 + 4 * cc;
        float acc = b[oc];
#pragma unroll
        for (int ky = 0; ky < 11; ++ky)
#pragma unroll
            for (int kx = 0; kx < 11; ++kx)
                acc += ip[ky * 75 + kx] * wp[ky * 11 + kx];
        conv_t[i] = fmaxf(acc, 0.f);
    }
    __syncthreads();

    for (int i = tid; i < 12 * 64; i += 256) {
        int oc = i / 64, py = (i / 8) % 8, px = i % 8;
        int gpy = py0 + py, gpx = px0 + px;
        if (gpy < P1H && gpx < P1W) {
            const float* cp = conv_t + oc * 289 + (2 * py) * 17 + 2 * px;
            float m = cp[0];
#pragma unroll
            for (int dy = 0; dy < 3; ++dy)
#pragma unroll
                for (int dx = 0; dx < 3; ++dx) m = fmaxf(m, cp[dy * 17 + dx]);
            out[((size_t)(f * 12 + oc) * P1H + gpy) * P1W + gpx] = m;
        }
    }
}

// ---------------- Kernel 2: conv2 5x5 s2 (12->24) + ReLU ----------------
__global__ void __launch_bounds__(256) k_conv2(const float* __restrict__ in,
                                               const float* __restrict__ w,
                                               const float* __restrict__ b,
                                               float* __restrict__ out) {
    __shared__ float w_s[24 * 12 * 25];
    const int tid = threadIdx.x;
    for (int i = tid; i < 24 * 12 * 25; i += 256) w_s[i] = w[i];
    __syncthreads();
    int idx = blockIdx.x * 256 + tid;
    const int total = TFRM * 24 * C2H * C2W;
    if (idx >= total) return;
    int x = idx % C2W, y = (idx / C2W) % C2H, oc = (idx / (C2W * C2H)) % 24, f = idx / (C2W * C2H * 24);
    const float* base = in + (size_t)f * 12 * P1H * P1W;
    float acc = b[oc];
    for (int ic = 0; ic < 12; ++ic) {
        const float* ip = base + ic * P1H * P1W + (2 * y) * P1W + 2 * x;
        const float* wp = w_s + (oc * 12 + ic) * 25;
#pragma unroll
        for (int ky = 0; ky < 5; ++ky)
#pragma unroll
            for (int kx = 0; kx < 5; ++kx) acc += ip[ky * P1W + kx] * wp[ky * 5 + kx];
    }
    out[idx] = fmaxf(acc, 0.f);
}

// ---------------- Generic 3x3 maxpool stride 2 ----------------
__global__ void __launch_bounds__(256) k_pool3s2(const float* __restrict__ in, float* __restrict__ out,
                                                 int Hi, int Wi, int Ho, int Wo, int total) {
    int idx = blockIdx.x * 256 + threadIdx.x;
    if (idx >= total) return;
    int x = idx % Wo, y = (idx / Wo) % Ho;
    int fc = idx / (Wo * Ho);
    const float* ip = in + (size_t)fc * Hi * Wi + (2 * y) * Wi + 2 * x;
    float m = ip[0];
#pragma unroll
    for (int dy = 0; dy < 3; ++dy)
#pragma unroll
        for (int dx = 0; dx < 3; ++dx) m = fmaxf(m, ip[dy * Wi + dx]);
    out[idx] = m;
}

// ---------------- Generic 3x3 s1 conv (24->24), optional ReLU ----------------
__global__ void __launch_bounds__(256) k_conv3x3(const float* __restrict__ in,
                                                 const float* __restrict__ w,
                                                 const float* __restrict__ b,
                                                 float* __restrict__ out,
                                                 int Hi, int Wi, int Ho, int Wo, int total, int do_relu) {
    __shared__ float w_s[24 * 24 * 9];
    const int tid = threadIdx.x;
    for (int i = tid; i < 24 * 24 * 9; i += 256) w_s[i] = w[i];
    __syncthreads();
    int idx = blockIdx.x * 256 + tid;
    if (idx >= total) return;
    int x = idx % Wo, y = (idx / Wo) % Ho, oc = (idx / (Wo * Ho)) % 24, f = idx / (Wo * Ho * 24);
    const float* base = in + (size_t)f * 24 * Hi * Wi;
    float acc = b[oc];
    for (int ic = 0; ic < 24; ++ic) {
        const float* ip = base + ic * Hi * Wi + y * Wi + x;
        const float* wp = w_s + (oc * 24 + ic) * 9;
#pragma unroll
        for (int ky = 0; ky < 3; ++ky)
#pragma unroll
            for (int kx = 0; kx < 3; ++kx) acc += ip[ky * Wi + kx] * wp[ky * 3 + kx];
    }
    out[idx] = do_relu ? fmaxf(acc, 0.f) : acc;
}

// ---------------- f32 -> f16 conversion ----------------
__global__ void __launch_bounds__(256) k_cvt_f16(const float* __restrict__ src,
                                                 _Float16* __restrict__ dst, int n) {
    int i = blockIdx.x * 256 + threadIdx.x;
    if (i < n) dst[i] = (_Float16)src[i];
}

// ---------------- WMMA input-projection GEMM -------------------------------
// proj[128,512] = feats[128,576] @ Wih^T[576,512] + (bih+bhh)
// A frag (16x32 f16): lane m=l&15, s=l>>4; halves 0..7 -> K=s*8+h, 8..15 -> 16+s*8+(h-8)
// B frag (32x16): column n=l&15, same K packing.  C/D f32: elem v -> row v+8*s, col l&15.
__global__ void __launch_bounds__(256) k_proj_wmma(const _Float16* __restrict__ A,
                                                   const _Float16* __restrict__ Bw,
                                                   const float* __restrict__ bih,
                                                   const float* __restrict__ bhh,
                                                   float* __restrict__ proj) {
    const int wave = (blockIdx.x * 256 + threadIdx.x) >> 5;   // 0..255
    const int lane = threadIdx.x & 31;
    const int mt = wave >> 5;      // 0..7
    const int nt = wave & 31;      // 0..31
    const int m = lane & 15, s = lane >> 4;
    v8f acc = {};
    for (int kt = 0; kt < 18; ++kt) {
        v16h a, bfr;
        const int kbase = kt * 32;
#pragma unroll
        for (int h = 0; h < 16; ++h) {
            int k = kbase + ((h < 8) ? (s * 8 + h) : (16 + s * 8 + (h - 8)));
            a[h]   = A[(mt * 16 + m) * FEAT + k];
            bfr[h] = Bw[(nt * 16 + m) * FEAT + k];
        }
        acc = __builtin_amdgcn_wmma_f32_16x16x32_f16(false, a, false, bfr,
                                                     (short)0, acc, false, false);
    }
    const int n = nt * 16 + m;
    const float bias = bih[n] + bhh[n];
#pragma unroll
    for (int v = 0; v < 8; ++v) {
        int mo = mt * 16 + v + 8 * s;
        proj[(size_t)mo * G4 + n] = acc[v] + bias;
    }
}

// ---------------- Persistent WMMA LSTM recurrence + FC + log_softmax -------------------
// One workgroup, 1024 threads = 32 waves. Wave w owns gate columns [16w,16w+16) and keeps
// its 4 B-fragments of Whh^T (f16) resident in VGPRs for all 738 steps. Per step:
//   A frag row 0 = h (f16, from LDS via two ds_load_b128 per K-tile), rows 1..15 = 0;
//   4 accumulating v_wmma_f32_16x16x32_f16; lanes 0..15 hold gates in acc[0].
__global__ void __launch_bounds__(1024) k_lstm_wmma(const float* __restrict__ proj,
                                                    const float* __restrict__ whh,
                                                    const float* __restrict__ fcw,
                                                    const float* __restrict__ fcb,
                                                    float* __restrict__ out) {
    __shared__ _Float16 h16[HID];    // hidden state, f16 (WMMA A operand precision)
    __shared__ float    c_s[HID];    // cell state, f32
    __shared__ float    g_s[G4];     // gate pre-activations
    __shared__ float    cls[8];
    const int tid  = threadIdx.x;
    const int wave = tid >> 5;                 // 0..31 == N tile
    const int lane = tid & 31;
    const int m = lane & 15, s = lane >> 4;
    const int n = wave * 16 + m;               // gate column produced by this lane (row 0)

    // Preload resident B fragments: B[k][n] = Whh[n][k], f32 -> f16, once.
    v16h bfrag[4];
#pragma unroll
    for (int kt = 0; kt < 4; ++kt)
#pragma unroll
        for (int h = 0; h < 16; ++h) {
            int k = kt * 32 + ((h < 8) ? (s * 8 + h) : (16 + s * 8 + (h - 8)));
            bfrag[kt][h] = (_Float16)whh[(size_t)n * HID + k];
        }

    if (tid < HID) { h16[tid] = (_Float16)0.f; c_s[tid] = 0.f; }
    __syncthreads();

    const v8h* hv = (const v8h*)h16;           // 16 aligned 8-half chunks of h

    for (int t = 0; t < NSTEP; ++t) {
        const int frame = t / 6 + t % 6;       // overlapping fps-windows, h/c carried through
        v8f acc = {};
#pragma unroll
        for (int kt = 0; kt < 4; ++kt) {
            // halves 0..7: K = kt*32 + s*8 .. +7 ; halves 8..15: K = kt*32+16+s*8 .. +7
            v8h lo = hv[kt * 4 + s];
            v8h hi = hv[kt * 4 + 2 + s];
            v16h a;
#pragma unroll
            for (int h = 0; h < 8; ++h) {
                a[h]     = (m == 0) ? lo[h] : (_Float16)0.f;
                a[h + 8] = (m == 0) ? hi[h] : (_Float16)0.f;
            }
            acc = __builtin_amdgcn_wmma_f32_16x16x32_f16(false, a, false, bfrag[kt],
                                                         (short)0, acc, false, false);
        }
        if (lane < 16) g_s[n] = acc[0] + proj[(size_t)frame * G4 + n];
        __syncthreads();
        if (tid < HID) {
            float ig = 1.f / (1.f + __expf(-g_s[tid]));
            float fg = 1.f / (1.f + __expf(-g_s[HID + tid]));
            float gg = tanhf(g_s[2 * HID + tid]);
            float og = 1.f / (1.f + __expf(-g_s[3 * HID + tid]));
            float c = fg * c_s[tid] + ig * gg;
            c_s[tid] = c;
            h16[tid] = (_Float16)(og * tanhf(c));
        }
        __syncthreads();
    }

    if (tid < 5) {
        float acc = fcb[tid];
        for (int k = 0; k < HID; ++k) acc += fcw[tid * HID + k] * (float)h16[k];
        cls[tid] = acc;
    }
    __syncthreads();
    if (tid == 0) {
        float mx = cls[0];
        for (int k = 1; k < 5; ++k) mx = fmaxf(mx, cls[k]);
        float ssum = 0.f;
        for (int k = 0; k < 5; ++k) ssum += __expf(cls[k] - mx);
        float lse = mx + __logf(ssum);
        for (int k = 0; k < 5; ++k) out[k] = cls[k] - lse;
    }
}

extern "C" void kernel_launch(void* const* d_in, const int* in_sizes, int n_in,
                              void* d_out, int out_size, void* d_ws, size_t ws_size,
                              hipStream_t stream) {
    (void)in_sizes; (void)n_in; (void)out_size; (void)ws_size;
    const float* video = (const float*)d_in[0];
    const float* c1w = (const float*)d_in[1];  const float* c1b = (const float*)d_in[2];
    const float* c2w = (const float*)d_in[3];  const float* c2b = (const float*)d_in[4];
    const float* c3w = (const float*)d_in[5];  const float* c3b = (const float*)d_in[6];
    const float* c4w = (const float*)d_in[7];  const float* c4b = (const float*)d_in[8];
    const float* wih = (const float*)d_in[9];  const float* whh = (const float*)d_in[10];
    const float* bih = (const float*)d_in[11]; const float* bhh = (const float*)d_in[12];
    const float* fcw = (const float*)d_in[13]; const float* fcb = (const float*)d_in[14];
    float* out = (float*)d_out;

    // Workspace layout (bytes, all 256B-aligned); ~48 MB total
    char* ws = (char*)d_ws;
    size_t o = 0;
    float* pool1 = (float*)(ws + o); o += (size_t)TFRM * 12 * P1H * P1W * 4;
    float* conv2 = (float*)(ws + o); o += (size_t)TFRM * 24 * C2H * C2W * 4;
    float* pool2 = (float*)(ws + o); o += (size_t)TFRM * 24 * P2H * P2W * 4;
    float* conv3 = (float*)(ws + o); o += (size_t)TFRM * 24 * C3H * C3W * 4;
    float* conv4 = (float*)(ws + o); o += (size_t)TFRM * 24 * C4H * C4W * 4;
    float* feats = (float*)(ws + o); o += (size_t)TFRM * FEAT * 4;
    _Float16* featsH = (_Float16*)(ws + o); o += (size_t)TFRM * FEAT * 2;
    _Float16* wihH   = (_Float16*)(ws + o); o += (size_t)G4 * FEAT * 2;
    float* proj  = (float*)(ws + o); o += (size_t)TFRM * G4 * 4;

    // 1) fused conv1 + ReLU + pool1
    k_conv1_pool1<<<dim3((P1W + 7) / 8, (P1H + 7) / 8, TFRM), 256, 0, stream>>>(video, c1w, c1b, pool1);
    // 2) conv2 + ReLU
    {
        int total = TFRM * 24 * C2H * C2W;
        k_conv2<<<(total + 255) / 256, 256, 0, stream>>>(pool1, c2w, c2b, conv2);
    }
    // 3) pool2
    {
        int total = TFRM * 24 * P2H * P2W;
        k_pool3s2<<<(total + 255) / 256, 256, 0, stream>>>(conv2, pool2, C2H, C2W, P2H, P2W, total);
    }
    // 4) conv3 (no ReLU)
    {
        int total = TFRM * 24 * C3H * C3W;
        k_conv3x3<<<(total + 255) / 256, 256, 0, stream>>>(pool2, c3w, c3b, conv3,
                                                           P2H, P2W, C3H, C3W, total, 0);
    }
    // 5) conv4 + ReLU
    {
        int total = TFRM * 24 * C4H * C4W;
        k_conv3x3<<<(total + 255) / 256, 256, 0, stream>>>(conv3, c4w, c4b, conv4,
                                                           C3H, C3W, C4H, C4W, total, 1);
    }
    // 6) pool3 -> feats [128,576]
    {
        int total = TFRM * FEAT;
        k_pool3s2<<<(total + 255) / 256, 256, 0, stream>>>(conv4, feats, C4H, C4W, 4, 6, total);
    }
    // 7) f32 -> f16 operands
    k_cvt_f16<<<(TFRM * FEAT + 255) / 256, 256, 0, stream>>>(feats, featsH, TFRM * FEAT);
    k_cvt_f16<<<(G4 * FEAT + 255) / 256, 256, 0, stream>>>(wih, wihH, G4 * FEAT);
    // 8) WMMA projection GEMM (256 waves)
    k_proj_wmma<<<32, 256, 0, stream>>>(featsH, wihH, bih, bhh, proj);
    // 9) persistent WMMA recurrence (738 steps, Whh resident in VGPRs) + FC + log_softmax
    k_lstm_wmma<<<1, 1024, 0, stream>>>(proj, whh, fcw, fcb, out);
}
